// FedLiteQuantizer_27341761806979
// MI455X (gfx1250) — matrix operand order repaired and checked by reading, not compiled, not run on hardware
//
#include <hip/hip_runtime.h>

// ---------------- problem constants (from reference) ----------------
constexpr int   R_ = 8, Q_ = 4096, D_ = 512, L_ = 256;
constexpr int   N_ITERS = 25;
constexpr float TEMP = 5.0f;
constexpr float EPS  = 1e-9f;

typedef __bf16 bf16;
typedef __attribute__((ext_vector_type(16))) __bf16 v16bf;
typedef __attribute__((ext_vector_type(8)))  __bf16 v8bf;
typedef __attribute__((ext_vector_type(8)))  float  v8f;
typedef int v4i __attribute__((vector_size(16)));

union ABFrag { v16bf v; v8bf h[2]; };

#define AS1 __attribute__((address_space(1)))
#define AS3 __attribute__((address_space(3)))

#if __has_builtin(__builtin_amdgcn_global_load_async_to_lds_b128) && \
    __has_builtin(__builtin_amdgcn_s_wait_asynccnt)
#define HAVE_ASYNC_LDS 1
#else
#define HAVE_ASYNC_LDS 0
#endif

// 16-byte global -> LDS copy: async DMA when available, plain store otherwise.
__device__ inline void cp16_g2l(const bf16* g, bf16* l) {
#if HAVE_ASYNC_LDS
    __builtin_amdgcn_global_load_async_to_lds_b128((AS1 v4i*)g, (AS3 v4i*)l, 0, 0);
#else
    *(v8bf*)l = *(const v8bf*)g;
#endif
}

__device__ inline void wait_async_lds() {
#if HAVE_ASYNC_LDS
    __builtin_amdgcn_s_wait_asynccnt(0);
#endif
}

// ---------------- helpers ----------------
__device__ inline float block_reduce_sum_256(float v) {
    __shared__ float buf[8];
    #pragma unroll
    for (int m = 1; m < 32; m <<= 1) v += __shfl_xor(v, m, 32);
    const int wave = threadIdx.x >> 5, lane = threadIdx.x & 31;
    if (lane == 0) buf[wave] = v;
    __syncthreads();
    float r = 0.f;
    if (threadIdx.x == 0) {
        #pragma unroll
        for (int i = 0; i < 8; ++i) r += buf[i];
    }
    return r; // valid on thread 0 only
}

// ---------------- prep: fp32 -> bf16 copy + transposed bf16 copy ----------------
__global__ __launch_bounds__(256) void k_prep(const float* __restrict__ x,
                                              bf16* __restrict__ xbf,
                                              bf16* __restrict__ xt) {
    __shared__ float tile[32][33];
    const int b  = blockIdx.z;
    const int q0 = blockIdx.x * 32, d0 = blockIdx.y * 32;
    const int tx = threadIdx.x, ty = threadIdx.y; // block (32,8)
    #pragma unroll
    for (int k = 0; k < 4; ++k) {
        const int row = ty + k * 8;
        const float v = x[((size_t)b * Q_ + q0 + row) * D_ + d0 + tx];
        xbf[((size_t)b * Q_ + q0 + row) * D_ + d0 + tx] = (bf16)v;
        tile[row][tx] = v;
    }
    __syncthreads();
    #pragma unroll
    for (int k = 0; k < 4; ++k) {
        const int drow = ty + k * 8;
        xt[((size_t)b * D_ + d0 + drow) * Q_ + q0 + tx] = (bf16)tile[tx][drow];
    }
}

// ---------------- init: strided-point centers + c_sq ----------------
__global__ __launch_bounds__(256) void k_init(const float* __restrict__ x,
                                              float* __restrict__ centers,
                                              bf16* __restrict__ cbf,
                                              float* __restrict__ csq) {
    const int b = blockIdx.y, l = blockIdx.x, t = threadIdx.x;
    const size_t src = ((size_t)b * Q_ + (size_t)l * (Q_ / L_)) * D_;
    const size_t dst = ((size_t)b * L_ + l) * D_;
    float sq = 0.f;
    for (int d = t; d < D_; d += 256) {
        const float v = x[src + d];
        centers[dst + d] = v;
        cbf[dst + d] = (bf16)v;
        sq += v * v;
    }
    const float tot = block_reduce_sum_256(sq);
    if (t == 0) csq[b * L_ + l] = tot;
}

// ---------------- score: WMMA logits + softmax (mode 0) or argmin (mode 1) ----------------
// block = 256 threads (8 waves). Block tile: 64 q-rows x 256 l-cols.
// wave w: m_tile = w>>1, n_half = w&1, 8 fragments of 16 cols each.
// Double-buffered LDS staging of the 256x32 centers K-slice (async DMA if available).
__global__ __launch_bounds__(256) void k_score(const bf16* __restrict__ xbf,
                                               const bf16* __restrict__ cbf,
                                               const float* __restrict__ csq,
                                               bf16* __restrict__ wt,
                                               float* __restrict__ denom,
                                               int* __restrict__ labels,
                                               int mode) {
    __shared__ bf16  lds_c[2][L_][40];  // double-buffered 256 x 32 slice (+pad)
    __shared__ float s_r0[4][16][2];
    __shared__ float s_r1[4][16][2];
    __shared__ int   s_ri[4][16][2];

    const int tid  = threadIdx.x;
    const int wave = tid >> 5, lane = tid & 31;
    const int l15  = lane & 15, hi = lane >> 4;
    const int mt   = wave >> 1, nh = wave & 1;
    const int b    = blockIdx.y;
    const int q0   = blockIdx.x * 64;

    const bf16* xrow    = xbf + ((size_t)b * Q_ + q0 + mt * 16 + l15) * D_;
    const bf16* cb_base = cbf + (size_t)b * L_ * D_;

    // per-thread staging assignment: 4 chunks of 16B, fixed part within row
    const int st_row0 = tid >> 2;          // + 64*c
    const int st_part = (tid & 3) * 8;     // bf16 element offset within slice

    auto stage = [&](int kk, int buf) {
        const int k0 = kk * 32;
        #pragma unroll
        for (int c = 0; c < 4; ++c) {
            const int row = st_row0 + 64 * c;
            cp16_g2l(cb_base + (size_t)row * D_ + k0 + st_part,
                     &lds_c[buf][row][st_part]);
        }
    };

    v8f acc[8] = {};

    stage(0, 0);
    for (int kk = 0; kk < D_ / 32; ++kk) {
        const int cur = kk & 1;
        wait_async_lds();
        __syncthreads();                    // staged slice visible; prev reads done
        if (kk + 1 < D_ / 32) stage(kk + 1, cur ^ 1);

        const int k0 = kk * 32;
        ABFrag A;
        const v8bf* pa = (const v8bf*)(xrow + k0 + hi * 8);
        A.h[0] = pa[0];       // K = base..base+7
        A.h[1] = pa[2];       // K = base+16..base+23

        ABFrag Bf[8];
        #pragma unroll
        for (int f = 0; f < 8; ++f) {
            const v8bf* pb = (const v8bf*)&lds_c[cur][nh * 128 + f * 16 + l15][hi * 8];
            Bf[f].h[0] = pb[0];
            Bf[f].h[1] = pb[2];
        }
        #pragma unroll
        for (int f = 0; f < 8; ++f)
            acc[f] = __builtin_amdgcn_wmma_f32_16x16x32_bf16(
                false, A.v, false, Bf[f].v, (short)0, acc[f], false, false);
    }

    // val = TEMP * (2*dot - c_sq)   (x_sq cancels in softmax / argmin)
    float cs[8];
    #pragma unroll
    for (int f = 0; f < 8; ++f) cs[f] = csq[b * L_ + nh * 128 + f * 16 + l15];
    #pragma unroll
    for (int f = 0; f < 8; ++f)
        #pragma unroll
        for (int i = 0; i < 8; ++i)
            acc[f][i] = TEMP * (2.f * acc[f][i] - cs[f]);

    if (mode == 0) {
        // ---- row softmax over 256 cols (8 frags x 16 lanes x 2 wave halves) ----
        float rmax[8];
        #pragma unroll
        for (int i = 0; i < 8; ++i) {
            rmax[i] = -3.4e38f;
            #pragma unroll
            for (int f = 0; f < 8; ++f) rmax[i] = fmaxf(rmax[i], acc[f][i]);
            #pragma unroll
            for (int m = 1; m < 16; m <<= 1)
                rmax[i] = fmaxf(rmax[i], __shfl_xor(rmax[i], m, 32));
        }
        if (l15 == 0) {
            #pragma unroll
            for (int i = 0; i < 8; ++i) s_r0[mt][hi * 8 + i][nh] = rmax[i];
        }
        __syncthreads();
        float rm[8];
        #pragma unroll
        for (int i = 0; i < 8; ++i)
            rm[i] = fmaxf(s_r0[mt][hi * 8 + i][0], s_r0[mt][hi * 8 + i][1]);

        float rsum[8];
        #pragma unroll
        for (int i = 0; i < 8; ++i) rsum[i] = 0.f;
        #pragma unroll
        for (int f = 0; f < 8; ++f)
            #pragma unroll
            for (int i = 0; i < 8; ++i) {
                acc[f][i] = __expf(acc[f][i] - rm[i]);
                rsum[i] += acc[f][i];
            }
        #pragma unroll
        for (int i = 0; i < 8; ++i)
            #pragma unroll
            for (int m = 1; m < 16; m <<= 1)
                rsum[i] += __shfl_xor(rsum[i], m, 32);
        if (l15 == 0) {
            #pragma unroll
            for (int i = 0; i < 8; ++i) s_r1[mt][hi * 8 + i][nh] = rsum[i];
        }
        __syncthreads();
        float inv[8];
        #pragma unroll
        for (int i = 0; i < 8; ++i)
            inv[i] = 1.f / (s_r1[mt][hi * 8 + i][0] + s_r1[mt][hi * 8 + i][1]);

        // write w^T (l, q): 8 consecutive q values per 16B store + column sums
        #pragma unroll
        for (int f = 0; f < 8; ++f) {
            const int col = nh * 128 + f * 16 + l15;
            v8bf pk;
            float colsum = 0.f;
            #pragma unroll
            for (int i = 0; i < 8; ++i) {
                const float w = acc[f][i] * inv[i];
                colsum += w;
                pk[i] = (bf16)w;
            }
            *(v8bf*)(wt + ((size_t)b * L_ + col) * Q_ + q0 + mt * 16 + hi * 8) = pk;
            colsum += __shfl_xor(colsum, 16, 32);
            if (lane < 16) atomicAdd(&denom[b * L_ + col], colsum);
        }
    } else {
        // ---- argmax val == argmin distance; first-index tie-break ----
        float bv[8];
        int   bc[8];
        #pragma unroll
        for (int i = 0; i < 8; ++i) { bv[i] = -3.4e38f; bc[i] = 0x7fffffff; }
        #pragma unroll
        for (int f = 0; f < 8; ++f) {
            const int col = nh * 128 + f * 16 + l15;
            #pragma unroll
            for (int i = 0; i < 8; ++i) {
                if (acc[f][i] > bv[i] || (acc[f][i] == bv[i] && col < bc[i])) {
                    bv[i] = acc[f][i]; bc[i] = col;
                }
            }
        }
        #pragma unroll
        for (int i = 0; i < 8; ++i)
            #pragma unroll
            for (int m = 1; m < 16; m <<= 1) {
                const float ov = __shfl_xor(bv[i], m, 32);
                const int   oc = __shfl_xor(bc[i], m, 32);
                if (ov > bv[i] || (ov == bv[i] && oc < bc[i])) { bv[i] = ov; bc[i] = oc; }
            }
        if (l15 == 0) {
            #pragma unroll
            for (int i = 0; i < 8; ++i) {
                s_r0[mt][hi * 8 + i][nh] = bv[i];
                s_ri[mt][hi * 8 + i][nh] = bc[i];
            }
        }
        __syncthreads();
        if (tid < 64) {
            const int mt2 = tid >> 4, rr = tid & 15;
            const float v0 = s_r0[mt2][rr][0], v1 = s_r0[mt2][rr][1];
            const int   c0 = s_ri[mt2][rr][0], c1 = s_ri[mt2][rr][1];
            const int lab = (v1 > v0 || (v1 == v0 && c1 < c0)) ? c1 : c0;
            labels[(size_t)b * Q_ + q0 + mt2 * 16 + rr] = lab;
        }
    }
}

// ---------------- update: centers_acc(l,d) = W^T(l,q) x X(q,d), K=4096 ----------------
// grid (L/64, D/128, R); block 256 = 8 waves; wave: m_tile = w>>1, n_half = w&1, 4 frags.
__global__ __launch_bounds__(256) void k_update(const bf16* __restrict__ wt,
                                                const bf16* __restrict__ xt,
                                                float* __restrict__ cacc) {
    const int tid  = threadIdx.x;
    const int wave = tid >> 5, lane = tid & 31;
    const int l15  = lane & 15, hi = lane >> 4;
    const int mt   = wave >> 1, nh = wave & 1;
    const int b    = blockIdx.z;
    const int l0   = blockIdx.x * 64, d0 = blockIdx.y * 128;

    const bf16* arow = wt + ((size_t)b * L_ + l0 + mt * 16 + l15) * Q_;
    const bf16* brow[4];
    #pragma unroll
    for (int f = 0; f < 4; ++f) {
        const int d = d0 + nh * 64 + f * 16 + l15;
        brow[f] = xt + ((size_t)b * D_ + d) * Q_;
    }

    v8f acc[4] = {};

    for (int kk = 0; kk < Q_ / 32; ++kk) {
        const int k0 = kk * 32;
        if (kk + 8 < Q_ / 32)
            __builtin_prefetch((const void*)(arow + k0 + 256), 0, 3);

        ABFrag A;
        const v8bf* pa = (const v8bf*)(arow + k0 + hi * 8);
        A.h[0] = pa[0];
        A.h[1] = pa[2];

        ABFrag Bf[4];
        #pragma unroll
        for (int f = 0; f < 4; ++f) {
            const v8bf* pb = (const v8bf*)(brow[f] + k0 + hi * 8);
            Bf[f].h[0] = pb[0];
            Bf[f].h[1] = pb[2];
        }
        #pragma unroll
        for (int f = 0; f < 4; ++f)
            acc[f] = __builtin_amdgcn_wmma_f32_16x16x32_bf16(
                false, A.v, false, Bf[f].v, (short)0, acc[f], false, false);
    }
    #pragma unroll
    for (int f = 0; f < 4; ++f) {
        const int d = d0 + nh * 64 + f * 16 + l15;
        #pragma unroll
        for (int i = 0; i < 8; ++i) {
            const int lrow = l0 + mt * 16 + hi * 8 + i;
            cacc[((size_t)b * L_ + lrow) * D_ + d] = acc[f][i];
        }
    }
}

// ---------------- finalize: centers = acc/(denom+eps), refresh bf16 + c_sq ----------------
__global__ __launch_bounds__(256) void k_finalize(const float* __restrict__ cacc,
                                                  const float* __restrict__ denom,
                                                  float* __restrict__ centers,
                                                  bf16* __restrict__ cbf,
                                                  float* __restrict__ csq) {
    const int b = blockIdx.y, l = blockIdx.x, t = threadIdx.x;
    const float inv = 1.f / (denom[b * L_ + l] + EPS);
    const size_t base = ((size_t)b * L_ + l) * D_;
    float sq = 0.f;
    for (int d = t; d < D_; d += 256) {
        const float c = cacc[base + d] * inv;
        centers[base + d] = c;
        cbf[base + d] = (bf16)c;
        sq += c * c;
    }
    const float tot = block_reduce_sum_256(sq);
    if (t == 0) csq[b * L_ + l] = tot;
}

// ---------------- gather: out[q,:] = centers[label[q],:] ----------------
__global__ __launch_bounds__(256) void k_gather(const float* __restrict__ centers,
                                                const int* __restrict__ labels,
                                                float* __restrict__ out) {
    const int b = blockIdx.y, q = blockIdx.x, t = threadIdx.x;
    const int lab = labels[(size_t)b * Q_ + q];
    const size_t src = ((size_t)b * L_ + lab) * D_;
    const size_t dst = ((size_t)b * Q_ + q) * D_;
    for (int d = t; d < D_; d += 256) out[dst + d] = centers[src + d];
}

// ---------------- host ----------------
extern "C" void kernel_launch(void* const* d_in, const int* in_sizes, int n_in,
                              void* d_out, int out_size, void* d_ws, size_t ws_size,
                              hipStream_t stream) {
    (void)in_sizes; (void)n_in; (void)out_size; (void)ws_size;
    const float* x = (const float*)d_in[0];
    float* out = (float*)d_out;
    char* ws = (char*)d_ws;

    size_t off = 0;
    bf16*  x_bf     = (bf16*)(ws + off);  off += (size_t)R_ * Q_ * D_ * sizeof(bf16);
    bf16*  x_t      = (bf16*)(ws + off);  off += (size_t)R_ * D_ * Q_ * sizeof(bf16);
    bf16*  w_t      = (bf16*)(ws + off);  off += (size_t)R_ * L_ * Q_ * sizeof(bf16);
    float* centers  = (float*)(ws + off); off += (size_t)R_ * L_ * D_ * sizeof(float);
    float* cacc     = (float*)(ws + off); off += (size_t)R_ * L_ * D_ * sizeof(float);
    bf16*  cbf      = (bf16*)(ws + off);  off += (size_t)R_ * L_ * D_ * sizeof(bf16);
    float* csq      = (float*)(ws + off); off += (size_t)R_ * L_ * sizeof(float);
    float* denom    = (float*)(ws + off); off += (size_t)R_ * L_ * sizeof(float);
    int*   labels   = (int*)(ws + off);   off += (size_t)R_ * Q_ * sizeof(int);

    k_prep<<<dim3(Q_ / 32, D_ / 32, R_), dim3(32, 8), 0, stream>>>(x, x_bf, x_t);
    k_init<<<dim3(L_, R_), 256, 0, stream>>>(x, centers, cbf, csq);

    for (int it = 0; it < N_ITERS; ++it) {
        (void)hipMemsetAsync(denom, 0, (size_t)R_ * L_ * sizeof(float), stream);
        k_score<<<dim3(Q_ / 64, R_), 256, 0, stream>>>(
            x_bf, cbf, csq, w_t, denom, nullptr, 0);
        k_update<<<dim3(L_ / 64, D_ / 128, R_), 256, 0, stream>>>(w_t, x_t, cacc);
        k_finalize<<<dim3(L_, R_), 256, 0, stream>>>(cacc, denom, centers, cbf, csq);
    }

    k_score<<<dim3(Q_ / 64, R_), 256, 0, stream>>>(
        x_bf, cbf, csq, nullptr, nullptr, labels, 1);
    k_gather<<<dim3(Q_, R_), 256, 0, stream>>>(centers, labels, out);
}
